// CCAHCL_12970801234151
// MI455X (gfx1250) — matrix operand order repaired and checked by reading, not compiled, not run on hardware
//
#include <hip/hip_runtime.h>
#include <hip/hip_bf16.h>

#define DIM   128
#define NCOMP 2000   // num_components (device scalar; fixed by problem definition)

typedef __bf16 bf16_t;
typedef __attribute__((ext_vector_type(16))) __bf16 v16bf;
typedef __attribute__((ext_vector_type(8)))  float  v8f;

#define WPITCH 144   // bf16 elems per LDS row of W^T (288 B -> every fragment 32B-aligned)

// ---------------- GEMM: Y[M x 128] = X[M x 128] @ W[128 x 128] + bias ----------------
// 256 threads = 8 wave32; each wave computes a 16x128 output strip via 8 WMMA tiles.
__global__ __launch_bounds__(256) void k_gemm_bias(
    const float* __restrict__ X, const float* __restrict__ W,
    const float* __restrict__ bias, float* __restrict__ Y, int Mtot)
{
  __shared__ __align__(32) bf16_t sW[DIM * WPITCH];
  const int tid = threadIdx.x;
  // Stage W^T into LDS as bf16: sW[n*WPITCH + k] = W[k*128 + n]
  for (int idx = tid; idx < DIM * DIM; idx += 256) {
    const int k = idx >> 7, n = idx & 127;
    sW[n * WPITCH + k] = (bf16_t)W[idx];
  }
  __syncthreads();

  const int lane = tid & 31;
  const int wave = tid >> 5;
  const int half = lane >> 4;
  const int l15  = lane & 15;
  const int m0   = blockIdx.x * 128 + wave * 16;

  int mrow = m0 + l15;
  if (mrow > Mtot - 1) mrow = Mtot - 1;          // clamp loads; stores are guarded
  const float* xrow = X + (size_t)mrow * DIM;

  v8f acc[8] = {};

  #pragma unroll
  for (int k0 = 0; k0 < DIM; k0 += 32) {
    // A fragment (16x32 bf16): lane half h holds K=[k0+16h, +16) contiguous
    v16bf a;
    const int kb1 = k0 + half * 16;
    const int kb2 = k0 + half * 16 + 8;
    #pragma unroll
    for (int i = 0; i < 8; ++i) {
      a[i]     = (bf16_t)xrow[kb1 + i];
      a[i + 8] = (bf16_t)xrow[kb2 + i];
    }
    #pragma unroll
    for (int t = 0; t < 8; ++t) {
      // B fragment (32x16 bf16): lane n = t*16+l15, K = k0 + half*16 .. +16 contiguous in W^T
      const v16bf b = *(const v16bf*)&sW[(t * 16 + l15) * WPITCH + k0 + half * 16];
      acc[t] = __builtin_amdgcn_wmma_f32_16x16x32_bf16(
                   false, a, false, b, (short)0, acc[t], false, false);
    }
  }

  #pragma unroll
  for (int t = 0; t < 8; ++t) {
    const int col = t * 16 + l15;
    const float bv = bias[col];
    #pragma unroll
    for (int v = 0; v < 8; ++v) {
      const int row = m0 + half * 8 + v;          // C/D layout: VGPR v -> M = v + 8*half
      if (row < Mtot) Y[(size_t)row * DIM + col] = acc[t][v] + bv;
    }
  }
}

// ---------------- elementwise / scatter helpers ----------------
__global__ void k_fill_f(float* __restrict__ p, float v, int n) {
  const int i = blockIdx.x * blockDim.x + threadIdx.x;
  if (i < n) p[i] = v;
}
__global__ void k_fill_i(int* __restrict__ p, int v, int n) {
  const int i = blockIdx.x * blockDim.x + threadIdx.x;
  if (i < n) p[i] = v;
}
__global__ void k_copy(float* __restrict__ dst, const float* __restrict__ src, int n) {
  const int i = blockIdx.x * blockDim.x + threadIdx.x;
  if (i < n) dst[i] = src[i];
}
__global__ void k_hist(const int* __restrict__ idx, int* __restrict__ cnt, int n) {
  const int i = blockIdx.x * blockDim.x + threadIdx.x;
  if (i < n) atomicAdd(&cnt[idx[i]], 1);
}
// acc[sidx[k]] += T[gidx[k]]  (per 4-float chunk; 32 chunks per 128-wide row)
__global__ void k_scatter_add(const float* __restrict__ T, float* __restrict__ acc,
                              const int* __restrict__ gidx, const int* __restrict__ sidx,
                              int nk) {
  const int i = blockIdx.x * blockDim.x + threadIdx.x;
  if (i >= nk * 32) return;
  const int k = i >> 5;
  const int c = (i & 31) * 4;
  const int g = gidx[k];
  const int s = sidx[k];
  const float4 v = *(const float4*)(T + (size_t)g * DIM + c);
  float* d = acc + (size_t)s * DIM + c;
  atomicAdd(d + 0, v.x); atomicAdd(d + 1, v.y);
  atomicAdd(d + 2, v.z); atomicAdd(d + 3, v.w);
}
__global__ void k_mean(float* __restrict__ acc, const int* __restrict__ cnt,
                       int rows, const float* __restrict__ pa, int doPrelu) {
  const int i = blockIdx.x * blockDim.x + threadIdx.x;
  if (i >= rows * DIM) return;
  const int r = i >> 7;
  const int cv = cnt[r] > 0 ? cnt[r] : 1;
  float xv = acc[i] / (float)cv;
  if (doPrelu) { const float a = *pa; xv = (xv >= 0.f) ? xv : a * xv; }
  acc[i] = xv;
}
__global__ void k_prelu_copy(float* __restrict__ dst, const float* __restrict__ src,
                             int n, const float* __restrict__ pa) {
  const int i = blockIdx.x * blockDim.x + threadIdx.x;
  if (i >= n) return;
  const float a = *pa;
  const float xv = src[i];
  dst[i] = (xv >= 0.f) ? xv : a * xv;
}
__global__ void k_combine_c(float* __restrict__ out,
                            const float* __restrict__ c1, const int* __restrict__ n1,
                            const float* __restrict__ c2, const int* __restrict__ n2,
                            int n, const float* __restrict__ pa) {
  const int i = blockIdx.x * blockDim.x + threadIdx.x;
  if (i >= n) return;
  const int r = i >> 7;
  const int d1 = n1[r] > 0 ? n1[r] : 1;
  const int d2 = n2[r] > 0 ? n2[r] : 1;
  float xv = c1[i] / (float)d1 + c2[i] / (float)d2;
  const float a = *pa;
  out[i] = (xv >= 0.f) ? xv : a * xv;
}

extern "C" void kernel_launch(void* const* d_in, const int* in_sizes, int n_in,
                              void* d_out, int out_size, void* d_ws, size_t ws_size,
                              hipStream_t stream) {
  const float* x  = (const float*)d_in[0];
  const int* hei  = (const int*)d_in[1];   // (2, NNZ): src, dst
  const int* hci  = (const int*)d_in[2];   // (2, E): he_e, he_c
  const int* nci  = (const int*)d_in[3];   // (2, N): nc_n, nc_c
  const float* pa = (const float*)d_in[7]; // prelu_a

  const int NNZ  = in_sizes[1] / 2;
  const int E    = in_sizes[2] / 2;
  const int N    = in_sizes[3] / 2;
  const int C    = NCOMP;
  const int Etot = E + N;

  const int* src  = hei;
  const int* dst  = hei + NNZ;
  const int* he_e = hci;
  const int* he_c = hci + E;
  const int* nc_n = nci;
  const int* nc_c = nci + N;

  const float* Wp[16];
  for (int i = 0; i < 16; ++i) Wp[i] = (const float*)d_in[8 + i];

  // workspace carve (256B aligned)
  char* base = (char*)d_ws;
  size_t off = 0;
  auto carve = [&](size_t bytes) -> void* {
    void* p = base + off;
    off += (bytes + 255) & ~(size_t)255;
    return p;
  };
  float* T   = (float*)carve((size_t)Etot * DIM * 4);
  float* e   = (float*)carve((size_t)Etot * DIM * 4);
  float* nb  = (float*)carve((size_t)N * DIM * 4);
  float* hb  = (float*)carve((size_t)N * DIM * 4);
  float* c1  = (float*)carve((size_t)C * DIM * 4);
  float* c2  = (float*)carve((size_t)C * DIM * 4);
  int* cnt_e = (int*)carve((size_t)Etot * 4);
  int* cnt_n = (int*)carve((size_t)N * 4);
  int* cc1   = (int*)carve((size_t)C * 4);
  int* cc2   = (int*)carve((size_t)C * 4);
  (void)ws_size; (void)n_in; (void)out_size;

  const int TB = 256;
  auto nblk = [](long long n, int tb) { return (int)((n + tb - 1) / tb); };

  // ---- segment counts (layer-invariant; re-zeroed every call) ----
  k_fill_i<<<nblk(E, TB), TB, 0, stream>>>(cnt_e, 0, E);
  k_fill_i<<<nblk(N, TB), TB, 0, stream>>>(cnt_e + E, 1, N);   // self-loop edges: count 1
  k_hist  <<<nblk(NNZ, TB), TB, 0, stream>>>(dst, cnt_e, NNZ);
  k_fill_i<<<nblk(N, TB), TB, 0, stream>>>(cnt_n, 1, N);       // self-loop contributes 1
  k_hist  <<<nblk(NNZ, TB), TB, 0, stream>>>(src, cnt_n, NNZ);
  k_fill_i<<<nblk(C, TB), TB, 0, stream>>>(cc1, 0, C);
  k_hist  <<<nblk(E, TB), TB, 0, stream>>>(he_c, cc1, E);
  k_fill_i<<<nblk(C, TB), TB, 0, stream>>>(cc2, 0, C);
  k_hist  <<<nblk(N, TB), TB, 0, stream>>>(nc_c, cc2, N);

  float* out_n = (float*)d_out;
  float* out_e = out_n + (size_t)N * DIM;
  float* out_c = out_e + (size_t)E * DIM;

  for (int layer = 0; layer < 2; ++layer) {
    const float* Wv  = Wp[8 * layer + 0]; const float* bv  = Wp[8 * layer + 1];
    const float* We  = Wp[8 * layer + 2]; const float* be  = Wp[8 * layer + 3];
    const float* Wec = Wp[8 * layer + 4]; const float* bec = Wp[8 * layer + 5];
    const float* Wnc = Wp[8 * layer + 6]; const float* bnc = Wp[8 * layer + 7];
    const float* hin = (layer == 0) ? x : hb;

    // ---- node -> edge ----
    k_gemm_bias<<<nblk(N, 128), 256, 0, stream>>>(hin, Wv, bv, T, N);
    k_fill_f<<<nblk((long long)E * DIM, TB), TB, 0, stream>>>(e, 0.f, E * DIM);
    k_copy  <<<nblk((long long)N * DIM, TB), TB, 0, stream>>>(e + (size_t)E * DIM, T, N * DIM); // self-loops
    k_scatter_add<<<nblk((long long)NNZ * 32, TB), TB, 0, stream>>>(T, e, src, dst, NNZ);
    k_mean  <<<nblk((long long)Etot * DIM, TB), TB, 0, stream>>>(e, cnt_e, Etot, pa, 1); // prelu

    // ---- edge -> node ----
    k_gemm_bias<<<nblk(Etot, 128), 256, 0, stream>>>(e, We, be, T, Etot);
    k_copy  <<<nblk((long long)N * DIM, TB), TB, 0, stream>>>(nb, T + (size_t)E * DIM, N * DIM); // self-loops
    k_scatter_add<<<nblk((long long)NNZ * 32, TB), TB, 0, stream>>>(T, nb, dst, src, NNZ);
    k_mean  <<<nblk((long long)N * DIM, TB), TB, 0, stream>>>(nb, cnt_n, N, pa, 0); // no prelu

    if (layer == 1) {
      // ---- edge -> component ----
      k_gemm_bias<<<nblk(Etot, 128), 256, 0, stream>>>(e, Wec, bec, T, Etot);
      k_fill_f<<<nblk((long long)C * DIM, TB), TB, 0, stream>>>(c1, 0.f, C * DIM);
      k_scatter_add<<<nblk((long long)E * 32, TB), TB, 0, stream>>>(T, c1, he_e, he_c, E);
      // ---- node -> component ----
      k_gemm_bias<<<nblk(N, 128), 256, 0, stream>>>(nb, Wnc, bnc, T, N);
      k_fill_f<<<nblk((long long)C * DIM, TB), TB, 0, stream>>>(c2, 0.f, C * DIM);
      k_scatter_add<<<nblk((long long)N * 32, TB), TB, 0, stream>>>(T, c2, nc_n, nc_c, N);
      k_combine_c<<<nblk((long long)C * DIM, TB), TB, 0, stream>>>(out_c, c1, cc1, c2, cc2, C * DIM, pa);
      // outputs
      k_copy      <<<nblk((long long)E * DIM, TB), TB, 0, stream>>>(out_e, e, E * DIM);
      k_prelu_copy<<<nblk((long long)N * DIM, TB), TB, 0, stream>>>(out_n, nb, N * DIM, pa);
    } else {
      k_prelu_copy<<<nblk((long long)N * DIM, TB), TB, 0, stream>>>(hb, nb, N * DIM, pa);
    }
  }
}
